// AttentionEncoder_43241730736686
// MI455X (gfx1250) — matrix operand order repaired and checked by reading, not compiled, not run on hardware
//
#include <hip/hip_runtime.h>
#include <math.h>

typedef __bf16 bf16_t;
typedef __attribute__((ext_vector_type(16))) __bf16 v16bf;
typedef __attribute__((ext_vector_type(8)))  float  v8f;

#define LNUM 4
#define CDIM 512
#define FDIM 2048
#define HNUM 8
#define HD   64
#define BDIM 4
#define TDIM 1024
#define RWIN 4
#define NR   9
#define NEG_INF (-1e30f)

// ---------------- bf16 helpers (explicit RNE) ------------------------------
static __device__ __forceinline__ float bf2f(bf16_t b) {
  union { unsigned short s; bf16_t b; } in; in.b = b;
  union { unsigned int u; float f; } out; out.u = ((unsigned int)in.s) << 16;
  return out.f;
}
static __device__ __forceinline__ bf16_t f2bf(float f) {
  union { float f; unsigned int u; } in; in.f = f;
  unsigned int r = in.u + 0x7FFFu + ((in.u >> 16) & 1u);
  union { unsigned short s; bf16_t b; } out; out.s = (unsigned short)(r >> 16);
  return out.b;
}

// ---------------- CDNA5 async global->LDS (ASYNCcnt tracked) ---------------
static __device__ __forceinline__ void async_b128_to_lds(const void* gptr, void* lptr) {
  unsigned lds = (unsigned)(size_t)lptr;  // low 32 bits of generic ptr = LDS offset
  asm volatile("global_load_async_to_lds_b128 %0, %1, off"
               :: "v"(lds), "v"(gptr) : "memory");
}
static __device__ __forceinline__ void wait_async0() {
  asm volatile("s_wait_asynccnt 0x0" ::: "memory");
}

// ---------------- WMMA fragment gathers from LDS ---------------------------
// A 16x32 bf16 (ISA 7.12.2): lanes0-15 row=l15, K=0..7 (V0-3), 16..23 (V4-7);
// lanes16-31 row=l15, K=8..15, 24..31.  `base` = row-major tile, k contiguous.
static __device__ __forceinline__ v16bf load_frag_a(const bf16_t* base, int rowstride) {
  int lane = threadIdx.x & 31;
  int hlf = lane >> 4, r = lane & 15;
  const unsigned int* p0 = (const unsigned int*)(base + r * rowstride + 8 * hlf);
  const unsigned int* p1 = (const unsigned int*)(base + r * rowstride + 16 + 8 * hlf);
  union { v16bf v; unsigned int u[8]; } f;
#pragma unroll
  for (int j = 0; j < 4; j++) f.u[j] = p0[j];
#pragma unroll
  for (int j = 0; j < 4; j++) f.u[4 + j] = p1[j];
  return f.v;
}
// B 32x16 bf16: lane n=l15; lanes0-15 K=0..7, 16..23; lanes16-31 K=8..15, 24..31.
// `base` = (n-major, k contiguous) tile.
static __device__ __forceinline__ v16bf load_frag_b(const bf16_t* base, int rowstride) {
  int lane = threadIdx.x & 31;
  int hlf = lane >> 4, n = lane & 15;
  const unsigned int* p = (const unsigned int*)(base + n * rowstride + 8 * hlf);
  const unsigned int* q = (const unsigned int*)(base + n * rowstride + 16 + 8 * hlf);
  union { v16bf v; unsigned int u[8]; } f;
#pragma unroll
  for (int j = 0; j < 4; j++) f.u[j] = p[j];
#pragma unroll
  for (int j = 0; j < 4; j++) f.u[4 + j] = q[j];
  return f.v;
}

// ---------------- f32 -> bf16 conversion (optional (b,t) mask) -------------
__global__ void k_convert(const float* __restrict__ src, bf16_t* __restrict__ dst,
                          size_t n, const float* __restrict__ mask, int Cd, int Td) {
  size_t i = (size_t)blockIdx.x * blockDim.x + threadIdx.x;
  if (i >= n) return;
  float v = src[i];
  if (mask) {
    size_t t = i % (size_t)Td;
    size_t b = i / ((size_t)Cd * Td);
    v *= mask[b * Td + t];
  }
  dst[i] = f2bf(v);
}

// ---------------- GEMM: out[b] = W(MxK) x X[b](KxN) + bias -----------------
// block tile 64x64, K-step 64, double-buffered LDS with async A staging.
// 8 waves: wave w -> M-tile (w&3), N-tiles 2*(w>>2)+{0,1}
__global__ __launch_bounds__(256) void k_gemm(
    const bf16_t* __restrict__ Wm, const bf16_t* __restrict__ Xb,
    const float* __restrict__ bias, float* __restrict__ outF,
    bf16_t* __restrict__ outB, const float* __restrict__ maskN,
    int M, int N, int K, float scale, int relu) {
  __shared__ bf16_t As[2][64 * 64];   // (m,k): async B128 staged, 128B rows
  __shared__ bf16_t Bst[2][64 * 66];  // (n,k): transposed at staging, padded
  int b = blockIdx.z;
  int n0 = blockIdx.x * 64, m0 = blockIdx.y * 64;
  int tid = threadIdx.x;
  int wave = tid >> 5, lane = tid & 31;
  int hlf = lane >> 4, l15 = lane & 15;
  int wm = wave & 3, wn = wave >> 2;
  const bf16_t* Xbb = Xb + (size_t)b * K * N;
  int crow = tid >> 3;           // 0..31 (k-row for B, m-row for A; +32 rep)
  int ccol = (tid & 7) * 8;      // 8-elem chunk column

  v8f zero = {};
  v8f acc[2]; acc[0] = zero; acc[1] = zero;

  auto stageA = [&](int bsel, int k0) {
#pragma unroll
    for (int rep = 0; rep < 2; rep++) {
      int row = crow + rep * 32;
      async_b128_to_lds(Wm + (size_t)(m0 + row) * K + k0 + ccol,
                        &As[bsel][row * 64 + ccol]);
    }
  };
  auto loadB = [&](int k0, uint4& r0, uint4& r1) {
    r0 = *(const uint4*)(Xbb + (size_t)(k0 + crow) * N + n0 + ccol);
    r1 = *(const uint4*)(Xbb + (size_t)(k0 + crow + 32) * N + n0 + ccol);
  };
  auto storeB = [&](int bsel, uint4 r0, uint4 r1) {
    union { uint4 q; unsigned short s[8]; } u;
    u.q = r0;
#pragma unroll
    for (int e = 0; e < 8; e++)
      ((unsigned short*)Bst[bsel])[(ccol + e) * 66 + crow] = u.s[e];
    u.q = r1;
#pragma unroll
    for (int e = 0; e < 8; e++)
      ((unsigned short*)Bst[bsel])[(ccol + e) * 66 + crow + 32] = u.s[e];
  };

  // prologue: stage tile 0
  stageA(0, 0);
  {
    uint4 r0, r1;
    loadB(0, r0, r1);
    storeB(0, r0, r1);
  }
  wait_async0();
  __syncthreads();

  int buf = 0;
  for (int k0 = 0; k0 < K; k0 += 64) {
    int nxt = buf ^ 1;
    int have = (k0 + 64 < K) ? 1 : 0;
    uint4 p0 = {0, 0, 0, 0}, p1 = {0, 0, 0, 0};
    if (have) {            // prefetch next tile: DMA A, global B -> regs
      stageA(nxt, k0 + 64);
      loadB(k0 + 64, p0, p1);
    }
    // compute current tile: 4 WMMAs per wave
    v16bf a0 = load_frag_a(&As[buf][wm * 16 * 64], 64);
    v16bf a1 = load_frag_a(&As[buf][wm * 16 * 64 + 32], 64);
#pragma unroll
    for (int nt = 0; nt < 2; nt++) {
      v16bf bb0 = load_frag_b(&Bst[buf][(wn * 2 + nt) * 16 * 66], 66);
      acc[nt] = __builtin_amdgcn_wmma_f32_16x16x32_bf16(
          false, a0, false, bb0, (short)0, acc[nt], false, false);
      v16bf bb1 = load_frag_b(&Bst[buf][(wn * 2 + nt) * 16 * 66 + 32], 66);
      acc[nt] = __builtin_amdgcn_wmma_f32_16x16x32_bf16(
          false, a1, false, bb1, (short)0, acc[nt], false, false);
    }
    if (have) storeB(nxt, p0, p1);
    wait_async0();
    __syncthreads();
    buf = nxt;
  }
  // epilogue: C layout VGPR r -> (M = r + 8*hlf, N = l15)
#pragma unroll
  for (int nt = 0; nt < 2; nt++) {
#pragma unroll
    for (int r = 0; r < 8; r++) {
      int m = m0 + wm * 16 + r + 8 * hlf;
      int n = n0 + (wn * 2 + nt) * 16 + l15;
      float v = acc[nt][r];
      if (bias) v += bias[m];
      v *= scale;
      if (relu) v = fmaxf(v, 0.f);
      if (maskN) v *= maskN[(size_t)b * N + n];
      size_t oidx = ((size_t)b * M + m) * N + n;
      if (outF) outF[oidx] = v;
      if (outB) outB[oidx] = f2bf(v);
    }
  }
}

// ---------------- band projections p[b,h,i,r] = q_i . emb_rel_k[r] ---------
__global__ void k_pband(const bf16_t* __restrict__ q, const float* __restrict__ ek,
                        float* __restrict__ pband) {
  size_t idx = (size_t)blockIdx.x * blockDim.x + threadIdx.x;
  size_t total = (size_t)BDIM * HNUM * TDIM * NR;
  if (idx >= total) return;
  int r = (int)(idx % NR);
  size_t bhi = idx / NR;
  int i = (int)(bhi % TDIM);
  size_t bh = bhi / TDIM;  // b*H + h
  const bf16_t* qp = q + (bh * HD) * TDIM + i;
  const float* ep = ek + (size_t)r * HD;
  float s = 0.f;
  for (int d = 0; d < HD; d++) s += bf2f(qp[(size_t)d * TDIM]) * ep[d];
  pband[idx] = s;
}

// ---------------- fused attention (flash-style streaming softmax) ----------
// grid (T/64, H, B), 128 threads = 4 waves, each wave owns 16 q rows.
__global__ __launch_bounds__(128) void k_attn(
    const bf16_t* __restrict__ q, const bf16_t* __restrict__ k,
    const bf16_t* __restrict__ v, const float* __restrict__ mask,
    const float* __restrict__ pband, const float* __restrict__ rv,
    bf16_t* __restrict__ obuf) {
  __shared__ bf16_t Qs[64 * 66];       // (t,d) padded
  __shared__ bf16_t Ks[64 * 66];       // (t,d) padded
  __shared__ bf16_t Vs[64 * 64];       // (d,t): async B128 staged
  __shared__ bf16_t Ps[4 * 16 * 66];   // per-wave probs (row, t_j) padded
  __shared__ float  Os[64 * 65];       // (t,d) output accum, padded
  __shared__ float  Ms[64], Ls[64];
  int b = blockIdx.z, h = blockIdx.y;
  int i0 = blockIdx.x * 64;
  int tid = threadIdx.x, wave = tid >> 5, lane = tid & 31;
  int hlf = lane >> 4, l15 = lane & 15;
  size_t chbase = ((size_t)b * HNUM + h) * HD;
  const bf16_t* qg = q + chbase * TDIM;
  const bf16_t* kg = k + chbase * TDIM;
  const bf16_t* vg = v + chbase * TDIM;

  // stage Q with transpose (d,t)->(t,d)
#pragma unroll
  for (int it = 0; it < 4; it++) {
    int idx = tid + it * 128;
    int d = idx >> 3, tc = (idx & 7) * 8;
    union { uint4 qd; unsigned short s[8]; } u;
    u.qd = *(const uint4*)(qg + (size_t)d * TDIM + i0 + tc);
#pragma unroll
    for (int e = 0; e < 8; e++)
      ((unsigned short*)Qs)[(tc + e) * 66 + d] = u.s[e];
  }

  v8f zero = {};
  v8f acc[4];
#pragma unroll
  for (int nt = 0; nt < 4; nt++) acc[nt] = zero;
  float mrow[8], lrow[8];
#pragma unroll
  for (int r = 0; r < 8; r++) { mrow[r] = NEG_INF; lrow[r] = 0.f; }

  for (int j0 = 0; j0 < TDIM; j0 += 64) {
    __syncthreads();
#pragma unroll
    for (int it = 0; it < 4; it++) {  // stage K (transpose) + V (async DMA)
      int idx = tid + it * 128;
      int d = idx >> 3, tc = (idx & 7) * 8;
      union { uint4 qd; unsigned short s[8]; } u;
      u.qd = *(const uint4*)(kg + (size_t)d * TDIM + j0 + tc);
#pragma unroll
      for (int e = 0; e < 8; e++)
        ((unsigned short*)Ks)[(tc + e) * 66 + d] = u.s[e];
      async_b128_to_lds(vg + (size_t)d * TDIM + j0 + tc, &Vs[d * 64 + tc]);
    }
    wait_async0();
    __syncthreads();

    // S = Q x K^T  (A from Qs, B from Ks: n=t_j, k=d contiguous)
    float s[4][8];
    v16bf a0 = load_frag_a(&Qs[wave * 16 * 66], 66);
    v16bf a1 = load_frag_a(&Qs[wave * 16 * 66 + 32], 66);
#pragma unroll
    for (int nt = 0; nt < 4; nt++) {
      v8f c = zero;
      v16bf b0 = load_frag_b(&Ks[nt * 16 * 66], 66);
      c = __builtin_amdgcn_wmma_f32_16x16x32_bf16(false, a0, false, b0, (short)0, c, false, false);
      v16bf b1 = load_frag_b(&Ks[nt * 16 * 66 + 32], 66);
      c = __builtin_amdgcn_wmma_f32_16x16x32_bf16(false, a1, false, b1, (short)0, c, false, false);
#pragma unroll
      for (int r = 0; r < 8; r++) s[nt][r] = c[r];
    }
    // relative-k band term + mask
#pragma unroll
    for (int nt = 0; nt < 4; nt++) {
      int j = j0 + nt * 16 + l15;
      float mv = mask[(size_t)b * TDIM + j];
#pragma unroll
      for (int r = 0; r < 8; r++) {
        int i = i0 + wave * 16 + r + 8 * hlf;
        int rel = j - i;
        float sv = s[nt][r];
        if (rel >= -RWIN && rel <= RWIN)
          sv += pband[(((size_t)b * HNUM + h) * TDIM + i) * NR + (rel + RWIN)];
        if (mv == 0.f) sv = NEG_INF;
        s[nt][r] = sv;
      }
    }
    // online softmax (rows live in 16-lane halves -> xor-shuffle 1,2,4,8)
#pragma unroll
    for (int r = 0; r < 8; r++) {
      float cm = fmaxf(fmaxf(s[0][r], s[1][r]), fmaxf(s[2][r], s[3][r]));
#pragma unroll
      for (int off = 1; off <= 8; off <<= 1) cm = fmaxf(cm, __shfl_xor(cm, off, 32));
      float mnew = fmaxf(mrow[r], cm);
      float alpha = __expf(mrow[r] - mnew);
      mrow[r] = mnew;
      float rs = 0.f;
#pragma unroll
      for (int nt = 0; nt < 4; nt++) {
        float p = __expf(s[nt][r] - mnew);
        s[nt][r] = p; rs += p;
      }
#pragma unroll
      for (int off = 1; off <= 8; off <<= 1) rs += __shfl_xor(rs, off, 32);
      lrow[r] = lrow[r] * alpha + rs;
#pragma unroll
      for (int nt = 0; nt < 4; nt++) acc[nt][r] = acc[nt][r] * alpha;
    }
    // stage P (C-layout -> row-major bf16) in per-wave LDS region
#pragma unroll
    for (int nt = 0; nt < 4; nt++)
#pragma unroll
      for (int r = 0; r < 8; r++)
        ((unsigned short*)Ps)[wave * 1056 + (r + 8 * hlf) * 66 + nt * 16 + l15] =
            __builtin_bit_cast(unsigned short, f2bf(s[nt][r]));
    asm volatile("s_wait_dscnt 0" ::: "memory");
    // O += P x V  (A from Ps, B from Vs: n=d, k=t_j contiguous)
    v16bf pa0 = load_frag_a(&Ps[wave * 1056], 66);
    v16bf pa1 = load_frag_a(&Ps[wave * 1056 + 32], 66);
#pragma unroll
    for (int nt = 0; nt < 4; nt++) {
      v16bf b0 = load_frag_b(&Vs[nt * 16 * 64], 64);
      acc[nt] = __builtin_amdgcn_wmma_f32_16x16x32_bf16(false, pa0, false, b0, (short)0, acc[nt], false, false);
      v16bf b1 = load_frag_b(&Vs[nt * 16 * 64 + 32], 64);
      acc[nt] = __builtin_amdgcn_wmma_f32_16x16x32_bf16(false, pa1, false, b1, (short)0, acc[nt], false, false);
    }
  }

  if (l15 == 0) {
#pragma unroll
    for (int r = 0; r < 8; r++) {
      Ms[wave * 16 + r + 8 * hlf] = mrow[r];
      Ls[wave * 16 + r + 8 * hlf] = lrow[r];
    }
  }
#pragma unroll
  for (int nt = 0; nt < 4; nt++)
#pragma unroll
    for (int r = 0; r < 8; r++)
      Os[(wave * 16 + r + 8 * hlf) * 65 + nt * 16 + l15] = acc[nt][r];
  __syncthreads();

  // relative-v band term (9 cols/row), then normalize by l
  if (tid < 64) {
    int i = i0 + tid;
    float mi = Ms[tid];
    float li = fmaxf(Ls[tid], 1e-20f);
    for (int r = 0; r < NR; r++) {
      int j = i + r - RWIN;
      if (j < 0 || j >= TDIM) continue;
      if (mask[(size_t)b * TDIM + j] == 0.f) continue;
      float sc = 0.f;
      for (int d = 0; d < HD; d++)
        sc += bf2f(Qs[tid * 66 + d]) * bf2f(kg[(size_t)d * TDIM + j]);
      sc += pband[(((size_t)b * HNUM + h) * TDIM + i) * NR + r];
      float a = __expf(sc - mi);
      const float* rvp = rv + (size_t)r * HD;
      for (int d = 0; d < HD; d++) Os[tid * 65 + d] += a * rvp[d];
    }
    float inv = 1.f / li;
    for (int d = 0; d < HD; d++) Os[tid * 65 + d] *= inv;
  }
  __syncthreads();
  // store transposed back to (B,C,T) bf16
#pragma unroll
  for (int it = 0; it < 32; it++) {
    int idx = tid + it * 128;
    int d = idx >> 6, t = idx & 63;
    obuf[(chbase + d) * TDIM + i0 + t] = f2bf(Os[t * 65 + d]);
  }
}

// ---------------- x = LN(x + add) [* mask], in place over channel dim ------
__global__ __launch_bounds__(256) void k_add_ln(
    float* __restrict__ x, const float* __restrict__ add,
    const float* __restrict__ mask, int postMask,
    const float* __restrict__ g, const float* __restrict__ bta) {
  int b = blockIdx.y;
  int wave = threadIdx.x >> 5, lane = threadIdx.x & 31;
  int t = blockIdx.x * 8 + wave;
  float mv = mask[(size_t)b * TDIM + t];
  float vals[16];
  float sum = 0.f, sq = 0.f;
#pragma unroll
  for (int kk = 0; kk < 16; kk++) {
    int c = lane + kk * 32;
    size_t idx = ((size_t)b * CDIM + c) * TDIM + t;
    float y = x[idx] + add[idx];
    vals[kk] = y; sum += y; sq += y * y;
  }
#pragma unroll
  for (int off = 1; off <= 16; off <<= 1) {
    sum += __shfl_xor(sum, off, 32);
    sq  += __shfl_xor(sq,  off, 32);
  }
  float mean = sum / (float)CDIM;
  float var = sq / (float)CDIM - mean * mean;
  float rstd = rsqrtf(var + 1e-6f);
#pragma unroll
  for (int kk = 0; kk < 16; kk++) {
    int c = lane + kk * 32;
    size_t idx = ((size_t)b * CDIM + c) * TDIM + t;
    float o = (vals[kk] - mean) * rstd * g[c] + bta[c];
    if (postMask) o *= mv;
    x[idx] = o;
  }
}

// ---------------- host orchestration ---------------------------------------
extern "C" void kernel_launch(void* const* d_in, const int* in_sizes, int n_in,
                              void* d_out, int out_size, void* d_ws, size_t ws_size,
                              hipStream_t stream) {
  (void)in_sizes; (void)n_in; (void)out_size; (void)ws_size;
  const float* x     = (const float*)d_in[0];
  const float* mask  = (const float*)d_in[1];
  const float* qw    = (const float*)d_in[2];
  const float* qbias = (const float*)d_in[3];
  const float* kw    = (const float*)d_in[4];
  const float* kbias = (const float*)d_in[5];
  const float* vw    = (const float*)d_in[6];
  const float* vbias = (const float*)d_in[7];
  const float* owp   = (const float*)d_in[8];
  const float* obias = (const float*)d_in[9];
  const float* erk   = (const float*)d_in[10];
  const float* erv   = (const float*)d_in[11];
  const float* ln1g  = (const float*)d_in[12];
  const float* ln1b  = (const float*)d_in[13];
  const float* w1    = (const float*)d_in[14];
  const float* b1p   = (const float*)d_in[15];
  const float* w2    = (const float*)d_in[16];
  const float* b2p   = (const float*)d_in[17];
  const float* ln2g  = (const float*)d_in[18];
  const float* ln2b  = (const float*)d_in[19];

  char* wsb = (char*)d_ws;
  size_t off = 0;
  auto alloc = [&](size_t bytes) -> char* {
    char* p = wsb + off;
    off += (bytes + 255) & ~(size_t)255;
    return p;
  };
  const size_t nBCT = (size_t)BDIM * CDIM * TDIM;
  const size_t wcc  = (size_t)CDIM * CDIM;
  const size_t wfc  = (size_t)FDIM * CDIM;

  bf16_t* wqb  = (bf16_t*)alloc(LNUM * wcc * 2);
  bf16_t* wkb  = (bf16_t*)alloc(LNUM * wcc * 2);
  bf16_t* wvb  = (bf16_t*)alloc(LNUM * wcc * 2);
  bf16_t* wob  = (bf16_t*)alloc(LNUM * wcc * 2);
  bf16_t* w1b  = (bf16_t*)alloc(LNUM * wfc * 2);
  bf16_t* w2b  = (bf16_t*)alloc(LNUM * wfc * 2);
  float*  xcur = (float*) alloc(nBCT * 4);
  bf16_t* xb   = (bf16_t*)alloc(nBCT * 2);
  bf16_t* qb16 = (bf16_t*)alloc(nBCT * 2);
  bf16_t* kb16 = (bf16_t*)alloc(nBCT * 2);
  bf16_t* vb16 = (bf16_t*)alloc(nBCT * 2);
  float*  pband= (float*) alloc((size_t)BDIM * HNUM * TDIM * NR * 4);
  bf16_t* obuf = (bf16_t*)alloc(nBCT * 2);
  float*  tmpf = (float*) alloc(nBCT * 4);
  bf16_t* hb16 = (bf16_t*)alloc((size_t)BDIM * FDIM * TDIM * 2);

  auto conv = [&](const float* s, bf16_t* d, size_t n) {
    k_convert<<<(unsigned)((n + 255) / 256), 256, 0, stream>>>(s, d, n, nullptr, CDIM, TDIM);
  };
  conv(qw, wqb, LNUM * wcc);
  conv(kw, wkb, LNUM * wcc);
  conv(vw, wvb, LNUM * wcc);
  conv(owp, wob, LNUM * wcc);
  conv(w1, w1b, LNUM * wfc);
  conv(w2, w2b, LNUM * wfc);
  hipMemcpyAsync(xcur, x, nBCT * 4, hipMemcpyDeviceToDevice, stream);

  dim3 gC(TDIM / 64, CDIM / 64, BDIM);
  dim3 gF(TDIM / 64, FDIM / 64, BDIM);
  dim3 gA(TDIM / 64, HNUM, BDIM);
  dim3 gL(TDIM / 8, BDIM);
  unsigned gP = (unsigned)(((size_t)BDIM * HNUM * TDIM * NR + 255) / 256);
  unsigned gX = (unsigned)((nBCT + 255) / 256);
  const float inv_scale = 0.125f;  // 1/sqrt(HD=64)

  for (int i = 0; i < LNUM; i++) {
    // q,k,v projections (bf16 out; q pre-scaled)
    k_convert<<<gX, 256, 0, stream>>>(xcur, xb, nBCT, nullptr, CDIM, TDIM);
    k_gemm<<<gC, 256, 0, stream>>>(wqb + (size_t)i * wcc, xb, qbias + (size_t)i * CDIM,
                                   nullptr, qb16, nullptr, CDIM, TDIM, CDIM, inv_scale, 0);
    k_gemm<<<gC, 256, 0, stream>>>(wkb + (size_t)i * wcc, xb, kbias + (size_t)i * CDIM,
                                   nullptr, kb16, nullptr, CDIM, TDIM, CDIM, 1.f, 0);
    k_gemm<<<gC, 256, 0, stream>>>(wvb + (size_t)i * wcc, xb, vbias + (size_t)i * CDIM,
                                   nullptr, vb16, nullptr, CDIM, TDIM, CDIM, 1.f, 0);
    k_pband<<<gP, 256, 0, stream>>>(qb16, erk + (size_t)i * NR * HD, pband);
    k_attn<<<gA, 128, 0, stream>>>(qb16, kb16, vb16, mask, pband,
                                   erv + (size_t)i * NR * HD, obuf);
    // output projection + residual + LN1
    k_gemm<<<gC, 256, 0, stream>>>(wob + (size_t)i * wcc, obuf, obias + (size_t)i * CDIM,
                                   tmpf, nullptr, nullptr, CDIM, TDIM, CDIM, 1.f, 0);
    k_add_ln<<<gL, 256, 0, stream>>>(xcur, tmpf, mask, 0,
                                     ln1g + (size_t)i * CDIM, ln1b + (size_t)i * CDIM);
    // FFN
    k_convert<<<gX, 256, 0, stream>>>(xcur, xb, nBCT, mask, CDIM, TDIM);
    k_gemm<<<gF, 256, 0, stream>>>(w1b + (size_t)i * wfc, xb, b1p + (size_t)i * FDIM,
                                   nullptr, hb16, mask, FDIM, TDIM, CDIM, 1.f, 1);
    k_gemm<<<gC, 256, 0, stream>>>(w2b + (size_t)i * wfc, hb16, b2p + (size_t)i * CDIM,
                                   tmpf, nullptr, mask, CDIM, TDIM, FDIM, 1.f, 0);
    k_add_ln<<<gL, 256, 0, stream>>>(xcur, tmpf, mask, 1,
                                     ln2g + (size_t)i * CDIM, ln2b + (size_t)i * CDIM);
  }
  hipMemcpyAsync(d_out, xcur, nBCT * 4, hipMemcpyDeviceToDevice, stream);
}
